// BaseModel_32255204393001
// MI455X (gfx1250) — matrix operand order repaired and checked by reading, not compiled, not run on hardware
//
#include <hip/hip_runtime.h>
#include <hip/hip_bf16.h>

// ---------------------------------------------------------------------------
// Problem constants (from reference)
// ---------------------------------------------------------------------------
#define BATCH 256
#define TSEED 120
#define TTOT  144      // T_SEED + PRED
#define PREDN 24
#define JN    15
#define HDIM  1024
#define DDIM  135
#define FREQN 61       // rfft bins of 120
#define DH    256      // head dim (HDIM / 4 heads)

typedef __attribute__((ext_vector_type(16))) __bf16 bf16x16;
typedef __attribute__((ext_vector_type(8)))  float  f32x8;

// ---------------------------------------------------------------------------
// Generic WMMA GEMM:  C[m,n] = act( sum_k A[m,k] * W[n,k] + bias[n] ) + Cadd[m,n]
// A: M x K fp32 (row stride lda), W: N x K fp32 (row stride ldw), C: M x N fp32.
// fp32 -> bf16 conversion while staging tiles in LDS; fp32 accumulation.
// Tile: 128(M) x 64(N) per 256-thread block, K stepped by 32.
// 8 waves laid out 4(M) x 2(N); each wave owns a 32x32 patch = 2x2 WMMA tiles.
// Double-buffered LDS with register prefetch: one barrier per K step; next
// tile's global loads overlap the current tile's WMMAs.
// Designed for M % 128 == 0, N % 64 == 0 (all call sites); K tail guarded.
// ---------------------------------------------------------------------------
#define BM 128
#define BN 64
#define BK 32
#define KROW 40   // BK + 8 pad (bf16 elems) to spread LDS banks

union bfpack4 { __bf16 h[4]; uint2 u; };

template <int ACT>
__launch_bounds__(256)
__global__ void k_gemm(const float* __restrict__ A, int lda,
                       const float* __restrict__ W, int ldw,
                       const float* __restrict__ bias,
                       const float* __restrict__ Cadd,
                       float* __restrict__ C,
                       int M, int N, int K)
{
    __shared__ __attribute__((aligned(16))) __bf16 sA[2][BM * KROW];
    __shared__ __attribute__((aligned(16))) __bf16 sB[2][BN * KROW];

    const int tid   = threadIdx.x;
    const int m0    = blockIdx.y * BM;
    const int n0    = blockIdx.x * BN;
    const int wave  = tid >> 5;
    const int lane  = tid & 31;
    const int wm    = (wave >> 1) * 32;   // 0,32,64,96
    const int wn    = (wave & 1) * 32;    // 0,32
    const int lrow  = lane & 15;
    const int lhalf = lane >> 4;

    // staging decomposition: thread -> (row, 4-wide k chunk)
    const int arow = tid >> 3;            // rows 0..31 (+32*i), 8 chunks/row
    const int acol = (tid & 7) * 4;       // k offset within tile
    const bool ldaV4 = ((lda & 3) == 0);
    const bool ldwV4 = ((ldw & 3) == 0);

    // per-thread source row pointers (advance by k0)
    const float* pA[4];
    #pragma unroll
    for (int i = 0; i < 4; ++i) pA[i] = A + (size_t)(m0 + arow + 32 * i) * lda + acol;
    const float* pW[2];
    #pragma unroll
    for (int i = 0; i < 2; ++i) pW[i] = W + (size_t)(n0 + arow + 32 * i) * ldw + acol;

    float4 ra[4], rw[2];                  // register staging for next tile

    auto loadRegs = [&](int k0) {
        if (k0 + BK <= K) {
            if (ldaV4) {
                #pragma unroll
                for (int i = 0; i < 4; ++i) ra[i] = *(const float4*)(pA[i] + k0);
            } else {
                #pragma unroll
                for (int i = 0; i < 4; ++i) {
                    const float* s = pA[i] + k0;
                    ra[i] = make_float4(s[0], s[1], s[2], s[3]);
                }
            }
            if (ldwV4) {
                #pragma unroll
                for (int i = 0; i < 2; ++i) rw[i] = *(const float4*)(pW[i] + k0);
            } else {
                #pragma unroll
                for (int i = 0; i < 2; ++i) {
                    const float* s = pW[i] + k0;
                    rw[i] = make_float4(s[0], s[1], s[2], s[3]);
                }
            }
        } else {
            // K tail (rare): guarded scalar loads
            int kb = k0 + acol;
            #pragma unroll
            for (int i = 0; i < 4; ++i) {
                const float* s = pA[i] + k0;
                ra[i] = make_float4(kb + 0 < K ? s[0] : 0.0f, kb + 1 < K ? s[1] : 0.0f,
                                    kb + 2 < K ? s[2] : 0.0f, kb + 3 < K ? s[3] : 0.0f);
            }
            #pragma unroll
            for (int i = 0; i < 2; ++i) {
                const float* s = pW[i] + k0;
                rw[i] = make_float4(kb + 0 < K ? s[0] : 0.0f, kb + 1 < K ? s[1] : 0.0f,
                                    kb + 2 < K ? s[2] : 0.0f, kb + 3 < K ? s[3] : 0.0f);
            }
        }
    };
    auto storeRegs = [&](int p) {
        #pragma unroll
        for (int i = 0; i < 4; ++i) {
            bfpack4 q;
            q.h[0] = (__bf16)ra[i].x; q.h[1] = (__bf16)ra[i].y;
            q.h[2] = (__bf16)ra[i].z; q.h[3] = (__bf16)ra[i].w;
            *(uint2*)&sA[p][(arow + 32 * i) * KROW + acol] = q.u;
        }
        #pragma unroll
        for (int i = 0; i < 2; ++i) {
            bfpack4 q;
            q.h[0] = (__bf16)rw[i].x; q.h[1] = (__bf16)rw[i].y;
            q.h[2] = (__bf16)rw[i].z; q.h[3] = (__bf16)rw[i].w;
            *(uint2*)&sB[p][(arow + 32 * i) * KROW + acol] = q.u;
        }
    };

    f32x8 acc[2][2] = {};

    // prologue: stage tile 0
    loadRegs(0);
    storeRegs(0);
    __syncthreads();

    int p = 0;
    for (int k0 = 0; k0 < K; k0 += BK) {
        const bool hasNext = (k0 + BK < K);
        if (hasNext) loadRegs(k0 + BK);   // global loads overlap the WMMAs below

        // A fragment (16x32 bf16, ISA 7.12.2): lane m=lane&15, half=lane>>4;
        // elems 0..7 -> k = half*8+e ; elems 8..15 -> k = 16+half*8+(e-8)
        bf16x16 afrag[2], bfrag[2];
        #pragma unroll
        for (int i = 0; i < 2; ++i) {
            const __bf16* pr = &sA[p][(wm + i * 16 + lrow) * KROW];
            ((uint4*)&afrag[i])[0] = *(const uint4*)(pr + lhalf * 8);
            ((uint4*)&afrag[i])[1] = *(const uint4*)(pr + 16 + lhalf * 8);
        }
        // B fragment (32x16 bf16): lane n=lane&15; elems 0..7 -> k = half*8+e ; elems 8..15 -> k = 16+half*8+(e-8)
        // B[k][n] = W[n][k]  (A @ W^T)
        #pragma unroll
        for (int j = 0; j < 2; ++j) {
            const __bf16* pr = &sB[p][(wn + j * 16 + lrow) * KROW];
            ((uint4*)&bfrag[j])[0] = *(const uint4*)(pr + lhalf * 8);
            ((uint4*)&bfrag[j])[1] = *(const uint4*)(pr + 16 + lhalf * 8);
        }
        #pragma unroll
        for (int i = 0; i < 2; ++i)
            #pragma unroll
            for (int j = 0; j < 2; ++j)
                acc[i][j] = __builtin_amdgcn_wmma_f32_16x16x32_bf16(
                    false, afrag[i], false, bfrag[j], (short)0, acc[i][j],
                    false, false);

        if (hasNext) {
            storeRegs(p ^ 1);             // fill the other buffer
            __syncthreads();              // single barrier per K step
            p ^= 1;
        }
    }

    // epilogue: C/D layout -> lane col = lane&15, row = (lane>>4)*8 + vgpr
    const bool fullTile = (m0 + BM <= M) && (n0 + BN <= N);
    if (fullTile) {
        #pragma unroll
        for (int i = 0; i < 2; ++i)
            #pragma unroll
            for (int j = 0; j < 2; ++j) {
                int n = n0 + wn + j * 16 + lrow;
                float bv = bias ? bias[n] : 0.0f;
                #pragma unroll
                for (int r = 0; r < 8; ++r) {
                    int m = m0 + wm + i * 16 + lhalf * 8 + r;
                    float x = acc[i][j][r] + bv;
                    if (ACT == 1) x = fmaxf(x, 0.0f);
                    if (Cadd) x += Cadd[(size_t)m * N + n];
                    C[(size_t)m * N + n] = x;
                }
            }
    } else {
        #pragma unroll
        for (int i = 0; i < 2; ++i)
            #pragma unroll
            for (int j = 0; j < 2; ++j) {
                int n = n0 + wn + j * 16 + lrow;
                if (n >= N) continue;
                float bv = bias ? bias[n] : 0.0f;
                #pragma unroll
                for (int r = 0; r < 8; ++r) {
                    int m = m0 + wm + i * 16 + lhalf * 8 + r;
                    if (m >= M) continue;
                    float x = acc[i][j][r] + bv;
                    if (ACT == 1) x = fmaxf(x, 0.0f);
                    if (Cadd) x += Cadd[(size_t)m * N + n];
                    C[(size_t)m * N + n] = x;
                }
            }
    }
}

// ---------------------------------------------------------------------------
// rfft(input_seq, axis=time).real == cosine DFT
// ---------------------------------------------------------------------------
__global__ void k_dft(const float* __restrict__ poses, float* __restrict__ freq)
{
    int f = blockIdx.x;   // 0..60
    int b = blockIdx.y;   // 0..255
    __shared__ float cosv[TSEED];
    float w = 6.28318530717958647692f * (float)f / (float)TSEED;
    for (int t = threadIdx.x; t < TSEED; t += blockDim.x)
        cosv[t] = __cosf(w * (float)t);
    __syncthreads();
    for (int d = threadIdx.x; d < DDIM; d += blockDim.x) {
        const float* p = poses + ((size_t)b * TTOT) * DDIM + d;
        float acc = 0.0f;
        #pragma unroll 4
        for (int t = 0; t < TSEED; ++t) acc += cosv[t] * p[(size_t)t * DDIM];
        freq[((size_t)b * FREQN + f) * DDIM + d] = acc;
    }
}

// ---------------------------------------------------------------------------
// Attention: one block per (query token, head, batch). qkv layout (B,61,3072)
// ---------------------------------------------------------------------------
__global__ void k_attn(const float* __restrict__ qkv, float* __restrict__ o)
{
    int tq = blockIdx.x, h = blockIdx.y, b = blockIdx.z;
    __shared__ float sc[64];
    const float* qrow = qkv + ((size_t)(b * FREQN + tq)) * (3 * HDIM) + h * DH;
    int wid = threadIdx.x >> 5, lane = threadIdx.x & 31;

    for (int k = wid; k < FREQN; k += 8) {
        const float* krow = qkv + ((size_t)(b * FREQN + k)) * (3 * HDIM) + HDIM + h * DH;
        float s = 0.0f;
        for (int d = lane; d < DH; d += 32) s += qrow[d] * krow[d];
        #pragma unroll
        for (int off = 16; off; off >>= 1) s += __shfl_xor(s, off, 32);
        if (lane == 0) sc[k] = s * (1.0f / 16.0f);   // 1/sqrt(256)
    }
    __syncthreads();

    float mx = -1e30f;
    for (int k = 0; k < FREQN; ++k) mx = fmaxf(mx, sc[k]);
    float sum = 0.0f;
    for (int k = 0; k < FREQN; ++k) sum += __expf(sc[k] - mx);
    float inv = 1.0f / sum;

    int d = threadIdx.x;   // 256 threads == DH
    float acc = 0.0f;
    for (int k = 0; k < FREQN; ++k) {
        float p = __expf(sc[k] - mx) * inv;
        acc += p * qkv[((size_t)(b * FREQN + k)) * (3 * HDIM) + 2 * HDIM + h * DH + d];
    }
    o[((size_t)(b * FREQN + tq)) * HDIM + h * DH + d] = acc;
}

// motion_ctx = mean over 61 tokens of projected attention output
__global__ void k_mean_ctx(const float* __restrict__ oproj, float* __restrict__ ctx)
{
    int idx = blockIdx.x * 256 + threadIdx.x;      // B*H
    int b = idx >> 10, j = idx & 1023;
    float s = 0.0f;
    for (int t = 0; t < FREQN; ++t) s += oproj[((size_t)(b * FREQN + t)) * HDIM + j];
    ctx[idx] = s * (1.0f / (float)FREQN);
}

// ---------------------------------------------------------------------------
// GRU gate fusion (in-place on h). gi/gh are (B, 3H) without biases.
// ---------------------------------------------------------------------------
__global__ void k_gru_fuse(const float* __restrict__ gi, const float* __restrict__ gh,
                           const float* __restrict__ bih, const float* __restrict__ bhh,
                           float* __restrict__ h, float* __restrict__ seqout, int seqStride)
{
    int idx = blockIdx.x * 256 + threadIdx.x;      // B*H
    int b = idx >> 10, j = idx & 1023;
    size_t base = (size_t)b * (3 * HDIM);
    float ir = gi[base + j]            + bih[j];
    float iz = gi[base + HDIM + j]     + bih[HDIM + j];
    float in = gi[base + 2 * HDIM + j] + bih[2 * HDIM + j];
    float hr = gh[base + j]            + bhh[j];
    float hz = gh[base + HDIM + j]     + bhh[HDIM + j];
    float hn = gh[base + 2 * HDIM + j] + bhh[2 * HDIM + j];
    float r = 1.0f / (1.0f + __expf(-(ir + hr)));
    float z = 1.0f / (1.0f + __expf(-(iz + hz)));
    float n = tanhf(in + r * hn);
    float hw = (1.0f - z) * n + z * h[idx];
    h[idx] = hw;
    if (seqout) seqout[(size_t)b * seqStride + j] = hw;
}

// SPL layer 2: delta[b,j,o] = sum_k z1[b,j*128+k] * w2[j,o,k] + b2[j,o]
__global__ void k_spl2(const float* __restrict__ z1, const float* __restrict__ w2,
                       const float* __restrict__ b2, float* __restrict__ delta)
{
    int idx = blockIdx.x * blockDim.x + threadIdx.x;   // B*J*6
    if (idx >= BATCH * JN * 6) return;
    int o = idx % 6, j = (idx / 6) % JN, b = idx / (JN * 6);
    const float* z = z1 + (size_t)b * (JN * 128) + j * 128;
    const float* w = w2 + ((size_t)j * 6 + o) * 128;
    float s = b2[j * 6 + o];
    #pragma unroll 4
    for (int k = 0; k < 128; ++k) s += z[k] * w[k];
    delta[idx] = s;
}

// init decode state: x0 = poses[:,119,:], prev6d = mat_to_6d(x0)
__global__ void k_init_decode(const float* __restrict__ poses,
                              float* __restrict__ prev6, float* __restrict__ xcur)
{
    int idx = blockIdx.x * blockDim.x + threadIdx.x;   // B*J
    if (idx >= BATCH * JN) return;
    int j = idx % JN, b = idx / JN;
    const float* x = poses + ((size_t)b * TTOT + (TSEED - 1)) * DDIM + j * 9;
    float* p = prev6 + (size_t)b * (JN * 6) + j * 6;
    p[0] = x[0]; p[1] = x[3]; p[2] = x[6];   // column 0
    p[3] = x[1]; p[4] = x[4]; p[5] = x[7];   // column 1
    float* xc = xcur + (size_t)b * DDIM + j * 9;
    #pragma unroll
    for (int i = 0; i < 9; ++i) xc[i] = x[i];
}

// rot6d step: new6d = prev6d + delta; emit pred6d[:,t,:]; Gram-Schmidt -> xcur
__global__ void k_rot6(const float* __restrict__ delta, float* __restrict__ prev6,
                       float* __restrict__ outT, float* __restrict__ xcur)
{
    int idx = blockIdx.x * blockDim.x + threadIdx.x;   // B*J
    if (idx >= BATCH * JN) return;
    int j = idx % JN, b = idx / JN;
    float v[6];
    #pragma unroll
    for (int i = 0; i < 6; ++i) {
        float nv = prev6[(size_t)b * 90 + j * 6 + i] + delta[(size_t)b * 90 + j * 6 + i];
        prev6[(size_t)b * 90 + j * 6 + i] = nv;
        outT[(size_t)b * (PREDN * 90) + j * 6 + i] = nv;
        v[i] = nv;
    }
    float a1x = v[0], a1y = v[2], a1z = v[4];
    float a2x = v[1], a2y = v[3], a2z = v[5];
    float n1 = fmaxf(sqrtf(a1x * a1x + a1y * a1y + a1z * a1z), 1e-12f);
    float b1x = a1x / n1, b1y = a1y / n1, b1z = a1z / n1;
    float d = b1x * a2x + b1y * a2y + b1z * a2z;
    float ox = a2x - d * b1x, oy = a2y - d * b1y, oz = a2z - d * b1z;
    float n2 = fmaxf(sqrtf(ox * ox + oy * oy + oz * oz), 1e-12f);
    float b2x = ox / n2, b2y = oy / n2, b2z = oz / n2;
    float b3x = b1y * b2z - b1z * b2y;
    float b3y = b1z * b2x - b1x * b2z;
    float b3z = b1x * b2y - b1y * b2x;
    float* xc = xcur + (size_t)b * DDIM + j * 9;
    xc[0] = b1x; xc[1] = b2x; xc[2] = b3x;
    xc[3] = b1y; xc[4] = b2y; xc[5] = b3y;
    xc[6] = b1z; xc[7] = b2z; xc[8] = b3z;
}

// ---------------------------------------------------------------------------
// Host orchestration
// ---------------------------------------------------------------------------
static inline void gemm(hipStream_t st, const float* A, int lda, const float* W, int ldw,
                        const float* bias, const float* Cadd, float* C,
                        int M, int N, int K, int act)
{
    dim3 g((N + BN - 1) / BN, (M + BM - 1) / BM);
    if (act) k_gemm<1><<<g, 256, 0, st>>>(A, lda, W, ldw, bias, Cadd, C, M, N, K);
    else     k_gemm<0><<<g, 256, 0, st>>>(A, lda, W, ldw, bias, Cadd, C, M, N, K);
}

extern "C" void kernel_launch(void* const* d_in, const int* in_sizes, int n_in,
                              void* d_out, int out_size, void* d_ws, size_t ws_size,
                              hipStream_t stream)
{
    (void)in_sizes; (void)n_in; (void)out_size; (void)ws_size;
    const float* poses      = (const float*)d_in[0];
    const float* freq_w     = (const float*)d_in[1];
    const float* freq_b     = (const float*)d_in[2];
    const float* attn_in_w  = (const float*)d_in[3];
    const float* attn_in_b  = (const float*)d_in[4];
    const float* attn_out_w = (const float*)d_in[5];
    const float* attn_out_b = (const float*)d_in[6];
    const float* gru_wih0   = (const float*)d_in[7];
    const float* gru_whh0   = (const float*)d_in[8];
    const float* gru_bih0   = (const float*)d_in[9];
    const float* gru_bhh0   = (const float*)d_in[10];
    const float* gru_wih1   = (const float*)d_in[11];
    const float* gru_whh1   = (const float*)d_in[12];
    const float* gru_bih1   = (const float*)d_in[13];
    const float* gru_bhh1   = (const float*)d_in[14];
    const float* pre_w      = (const float*)d_in[15];
    const float* pre_b      = (const float*)d_in[16];
    const float* spl_w1     = (const float*)d_in[17];  // (15,128,1024) == (1920,1024)
    const float* spl_b1     = (const float*)d_in[18];  // (1920,)
    const float* spl_w2     = (const float*)d_in[19];  // (15,6,128)
    const float* spl_b2     = (const float*)d_in[20];  // (15,6)
    float* out = (float*)d_out;                        // (B, 24, 90)

    // ---- workspace carving (~268 MB peak, regions reused across phases) ----
    char* base = (char*)d_ws;
    auto AL = [](size_t x) { return (x + 255) & ~(size_t)255; };
    const size_t szFreq = (size_t)BATCH * FREQN * DDIM * 4;      // 8.4 MB
    const size_t szFeat = (size_t)BATCH * FREQN * HDIM * 4;      // 64 MB
    const size_t szQkv  = (size_t)BATCH * FREQN * 3 * HDIM * 4;  // 192 MB
    const size_t oFreq = 0;
    const size_t oFeat = AL(oFreq + szFreq);
    const size_t oQkv  = AL(oFeat + szFeat);       // region R2 (reused later)
    const size_t oCtx  = AL(oQkv + szQkv);
    const size_t oH0   = AL(oCtx + (size_t)BATCH * HDIM * 4);
    const size_t oH1   = AL(oH0 + (size_t)BATCH * HDIM * 4);
    // aliases: attno overlays freq+feat; oproj + GRU-phase buffers overlay qkv
    const size_t oAttno = oFreq;
    const size_t oOproj = oQkv;
    const size_t oSeq0  = oQkv;                                    // 126 MB
    const size_t oGi    = AL(oSeq0 + (size_t)BATCH * TSEED * HDIM * 4);
    const size_t oGh    = AL(oGi + (size_t)BATCH * 3 * HDIM * 4);
    const size_t oHid   = AL(oGh + (size_t)BATCH * 3 * HDIM * 4);
    const size_t oZ1    = AL(oHid + (size_t)BATCH * HDIM * 4);
    const size_t oDelta = AL(oZ1 + (size_t)BATCH * JN * 128 * 4);
    const size_t oPrev  = AL(oDelta + (size_t)BATCH * 90 * 4);
    const size_t oXcur  = AL(oPrev + (size_t)BATCH * 90 * 4);

    float* freq  = (float*)(base + oFreq);
    float* feat  = (float*)(base + oFeat);
    float* qkv   = (float*)(base + oQkv);
    float* attno = (float*)(base + oAttno);
    float* oproj = (float*)(base + oOproj);
    float* ctx   = (float*)(base + oCtx);
    float* h0    = (float*)(base + oH0);
    float* h1    = (float*)(base + oH1);
    float* seq0  = (float*)(base + oSeq0);
    float* gi    = (float*)(base + oGi);
    float* gh    = (float*)(base + oGh);
    float* hid   = (float*)(base + oHid);
    float* z1    = (float*)(base + oZ1);
    float* delta = (float*)(base + oDelta);
    float* prev6 = (float*)(base + oPrev);
    float* xcur  = (float*)(base + oXcur);

    const int MF = BATCH * FREQN;   // 15616

    // ---- attention branch (fully parallel, large WMMA GEMMs) ----
    k_dft<<<dim3(FREQN, BATCH), 160, 0, stream>>>(poses, freq);
    gemm(stream, freq, DDIM, freq_w, DDIM, freq_b, nullptr, feat, MF, HDIM, DDIM, 0);
    gemm(stream, feat, HDIM, attn_in_w, HDIM, attn_in_b, nullptr, qkv, MF, 3 * HDIM, HDIM, 0);
    k_attn<<<dim3(FREQN, 4, BATCH), 256, 0, stream>>>(qkv, attno);
    gemm(stream, attno, HDIM, attn_out_w, HDIM, attn_out_b, nullptr, oproj, MF, HDIM, HDIM, 0);
    k_mean_ctx<<<(BATCH * HDIM) / 256, 256, 0, stream>>>(oproj, ctx);

    // ---- GRU layer 0 over seed sequence ----
    hipMemsetAsync(h0, 0, (size_t)BATCH * HDIM * 4, stream);
    for (int t = 0; t < TSEED; ++t) {
        gemm(stream, poses + (size_t)t * DDIM, TTOT * DDIM, gru_wih0, DDIM,
             nullptr, nullptr, gi, BATCH, 3 * HDIM, DDIM, 0);
        gemm(stream, h0, HDIM, gru_whh0, HDIM, nullptr, nullptr, gh, BATCH, 3 * HDIM, HDIM, 0);
        k_gru_fuse<<<(BATCH * HDIM) / 256, 256, 0, stream>>>(
            gi, gh, gru_bih0, gru_bhh0, h0, seq0 + (size_t)t * HDIM, TSEED * HDIM);
    }
    // ---- GRU layer 1 over layer-0 outputs ----
    hipMemsetAsync(h1, 0, (size_t)BATCH * HDIM * 4, stream);
    for (int t = 0; t < TSEED; ++t) {
        gemm(stream, seq0 + (size_t)t * HDIM, TSEED * HDIM, gru_wih1, HDIM,
             nullptr, nullptr, gi, BATCH, 3 * HDIM, HDIM, 0);
        gemm(stream, h1, HDIM, gru_whh1, HDIM, nullptr, nullptr, gh, BATCH, 3 * HDIM, HDIM, 0);
        k_gru_fuse<<<(BATCH * HDIM) / 256, 256, 0, stream>>>(
            gi, gh, gru_bih1, gru_bhh1, h1, nullptr, 0);
    }

    // ---- autoregressive decode ----
    k_init_decode<<<(BATCH * JN + 255) / 256, 256, 0, stream>>>(poses, prev6, xcur);
    for (int t = 0; t < PREDN; ++t) {
        gemm(stream, xcur, DDIM, gru_wih0, DDIM, nullptr, nullptr, gi, BATCH, 3 * HDIM, DDIM, 0);
        gemm(stream, h0, HDIM, gru_whh0, HDIM, nullptr, nullptr, gh, BATCH, 3 * HDIM, HDIM, 0);
        k_gru_fuse<<<(BATCH * HDIM) / 256, 256, 0, stream>>>(
            gi, gh, gru_bih0, gru_bhh0, h0, nullptr, 0);
        gemm(stream, h0, HDIM, gru_wih1, HDIM, nullptr, nullptr, gi, BATCH, 3 * HDIM, HDIM, 0);
        gemm(stream, h1, HDIM, gru_whh1, HDIM, nullptr, nullptr, gh, BATCH, 3 * HDIM, HDIM, 0);
        k_gru_fuse<<<(BATCH * HDIM) / 256, 256, 0, stream>>>(
            gi, gh, gru_bih1, gru_bhh1, h1, nullptr, 0);
        // hidden = relu(h1 @ pre_w^T + pre_b) + motion_ctx  (fused add)
        gemm(stream, h1, HDIM, pre_w, HDIM, pre_b, ctx, hid, BATCH, HDIM, HDIM, 1);
        // z1 = relu(hidden @ spl_w1^T + spl_b1), spl_w1 contiguous (1920,1024)
        gemm(stream, hid, HDIM, spl_w1, HDIM, spl_b1, nullptr, z1, BATCH, JN * 128, HDIM, 1);
        k_spl2<<<(BATCH * JN * 6 + 255) / 256, 256, 0, stream>>>(z1, spl_w2, spl_b2, delta);
        k_rot6<<<(BATCH * JN + 255) / 256, 256, 0, stream>>>(
            delta, prev6, out + (size_t)t * 90, xcur);
    }
}
